// ImpactFunNN_82609400971587
// MI455X (gfx1250) — compile-verified, matrix operand order checked by reading, not run on hardware
//
#include <hip/hip_runtime.h>

typedef __attribute__((ext_vector_type(16))) _Float16 v16h;
typedef __attribute__((ext_vector_type(8)))  _Float16 v8h;
typedef __attribute__((ext_vector_type(8)))  float    v8f;

#define BQ 4
#define SQ 256
#define DQ 16
#define HQ 128

// ---------------------------------------------------------------------------
// Prep 1: valid mask + relu'd embedding gathers, one thread per (b,s)
// ---------------------------------------------------------------------------
__global__ void prep_kernel(const float* __restrict__ ets,
                            const int* __restrict__ ind,
                            const float* __restrict__ emb1,
                            const float* __restrict__ emb2,
                            float* __restrict__ valid,
                            float* __restrict__ ee1,
                            float* __restrict__ ee2) {
  int s = blockIdx.x * blockDim.x + threadIdx.x;   // 0 .. B*S-1
  if (s >= BQ * SQ) return;
  const float* e = ets + (size_t)s * DQ;
  bool allz = true;
  for (int k = 0; k < DQ; ++k) allz = allz && (e[k] == 0.0f);
  valid[s] = allz ? 0.0f : 1.0f;
  int idx = ind[s];
  for (int d = 0; d < DQ; ++d) {
    ee1[s * DQ + d] = fmaxf(emb1[idx * DQ + d], 0.0f);
    ee2[s * DQ + d] = fmaxf(emb2[idx * DQ + d], 0.0f);
  }
}

// ---------------------------------------------------------------------------
// Prep 2: W2 [H x 256] f32 -> transposed f16 w2t[n][k], contiguous in k
// ---------------------------------------------------------------------------
__global__ void w2t_kernel(const float* __restrict__ w2, _Float16* __restrict__ w2t) {
  int t = blockIdx.x * blockDim.x + threadIdx.x;   // 0 .. 256*128-1
  if (t >= DQ * DQ * HQ) return;
  int n = t / HQ, k = t % HQ;
  w2t[t] = (_Float16)w2[k * (DQ * DQ) + n];
}

// Branch-free numerically-stable softplus: max(x,0) + log1p(exp(-|x|))
// Uses raw v_exp_f32 / v_log_f32 (base-2) via amdgcn builtins.
__device__ __forceinline__ float softplus_bf(float x) {
  const float LOG2E = 1.4426950408889634f;
  const float LN2   = 0.6931471805599453f;
  float p = __builtin_amdgcn_exp2f(-fabsf(x) * LOG2E);        // exp(-|x|) in (0,1]
  return fmaxf(x, 0.0f) + __builtin_amdgcn_logf(1.0f + p) * LN2;
}

// ---------------------------------------------------------------------------
// Main: one wave32 per 16-pair tile. fc2 via v_wmma_f32_16x16x32_f16,
// double-buffered B tiles so next-tile loads overlap the softplus epilogue.
// ---------------------------------------------------------------------------
__global__ __launch_bounds__(256) void impact_kernel(
    const float* __restrict__ tdm, const float* __restrict__ w1,
    const float* __restrict__ b1, const float* __restrict__ b2,
    const float* __restrict__ valid, const float* __restrict__ ee1,
    const float* __restrict__ ee2, const _Float16* __restrict__ w2t,
    float* __restrict__ out) {
  const int lane = threadIdx.x & 31;
  const int wave = threadIdx.x >> 5;
  const int tile = blockIdx.x * 8 + wave;          // 16384 tiles total
  const int p0   = tile * 16;
  const int b    = p0 / (SQ * SQ);
  const int rem  = p0 - b * SQ * SQ;
  const int i    = rem / SQ;
  const int j0   = rem % SQ;                       // multiple of 16
  const int row_i = b * SQ + i;

  const int e  = lane & 15;                        // column within 16-lane half
  const int hi = lane >> 4;                        // 0: lanes 0-15, 1: lanes 16-31

  // --- per-pair t = log(tdm * pos * pad_mask + 1), computed in lanes 0-15 ---
  float tlog;
  {
    int j = j0 + e;
    float raw = tdm[(size_t)row_i * SQ + j];
    float pos = raw > 0.0f ? 1.0f : 0.0f;
    float pm  = valid[row_i] * valid[b * SQ + j];
    tlog = __builtin_amdgcn_logf(raw * pos * pm + 1.0f) * 0.6931471805599453f;
  }
  float t_m = __shfl(tlog, e, 32);                 // t for row M = lane%16

  // --- build A = relu(t*w1+b1) in the ISA f16 16x32 A layout, 4 K-steps ----
  // lane<16 holds K {k0..k0+7, k0+16..k0+23}; lane>=16 the +8 shifted set.
  v16h A[4];
#pragma unroll
  for (int s4 = 0; s4 < 4; ++s4) {
    int kbase = s4 * 32 + hi * 8;
#pragma unroll
    for (int q = 0; q < 16; ++q) {
      int k = kbase + (q < 8 ? q : 8 + q);         // 0..7 then 16..23 (rel.)
      float h = fmaxf(fmaf(t_m, w1[k], b1[k]), 0.0f);
      A[s4][q] = (_Float16)h;
    }
  }

  // --- per-lane ee2 weights: C VGPR v holds row M = v + 8*hi, col e ---------
  float ee2v[8];
#pragma unroll
  for (int v = 0; v < 8; ++v) {
    int j = j0 + v + hi * 8;
    ee2v[v] = ee2[(size_t)(b * SQ + j) * DQ + e];
  }

  // B-row base for this lane's column (e); rows advance by 16*HQ per d-tile.
  const _Float16* bbase = w2t + (size_t)e * HQ;

  // Double-buffered B chunks: 8 x v8h per d-tile (4 K-steps x {lo,hi} run).
  v8h buf0[8], buf1[8];
#pragma unroll
  for (int s4 = 0; s4 < 4; ++s4) {
    const _Float16* p = bbase + s4 * 32 + hi * 8;
    buf0[2 * s4]     = *(const v8h*)(p);
    buf0[2 * s4 + 1] = *(const v8h*)(p + 16);
  }

  // acc[v] = sum_d ee1[d] * softplus(fc2[M, d*16+e])   (per-lane, no x-lane ops)
  float acc[8] = {0.f, 0.f, 0.f, 0.f, 0.f, 0.f, 0.f, 0.f};

#pragma unroll
  for (int d = 0; d < DQ; ++d) {                   // 16 N-tiles of fc2 output
    v8h* cur = (d & 1) ? buf1 : buf0;
    v8h* nxt = (d & 1) ? buf0 : buf1;

    // issue next tile's loads BEFORE consuming current tile -> overlap
    if (d + 1 < DQ) {
      const _Float16* brow = bbase + (size_t)(d + 1) * 16 * HQ;
#pragma unroll
      for (int s4 = 0; s4 < 4; ++s4) {
        const _Float16* p = brow + s4 * 32 + hi * 8;
        nxt[2 * s4]     = *(const v8h*)(p);
        nxt[2 * s4 + 1] = *(const v8h*)(p + 16);
      }
    }

    float b2v  = b2[d * 16 + e];
    float ee1d = ee1[(size_t)row_i * DQ + d];

    v8f c = {};
#pragma unroll
    for (int s4 = 0; s4 < 4; ++s4) {
      v16h bm;
#pragma unroll
      for (int q = 0; q < 8; ++q) {
        bm[q]     = cur[2 * s4][q];
        bm[8 + q] = cur[2 * s4 + 1][q];
      }
      c = __builtin_amdgcn_wmma_f32_16x16x32_f16(
              /*neg_a=*/false, A[s4], /*neg_b=*/false, bm,
              /*c_mod=*/(short)0, c, /*reuse_a=*/false, /*reuse_b=*/false);
    }

    // softplus epilogue co-executes with the in-flight next-tile loads
#pragma unroll
    for (int v = 0; v < 8; ++v) {
      float sp = softplus_bf(c[v] + b2v);          // branch-free
      acc[v] = fmaf(ee1d, sp, acc[v]);             // ee2 hoisted out of loop
    }
  }

  // --- fold ee2, then ONE cross-lane reduction over e per C register -------
  float red[8];
#pragma unroll
  for (int v = 0; v < 8; ++v) {
    float w = acc[v] * ee2v[v];
    w += __shfl_xor(w, 1, 16);
    w += __shfl_xor(w, 2, 16);
    w += __shfl_xor(w, 4, 16);
    w += __shfl_xor(w, 8, 16);
    red[v] = w;                                    // row sum, replicated in half
  }

  // --- final mask * pos, stored by lane 0 (M=0..7) and lane 16 (M=8..15) ----
  if (e == 0) {
#pragma unroll
    for (int v = 0; v < 8; ++v) {
      int j = j0 + v + hi * 8;
      float raw = tdm[(size_t)row_i * SQ + j];
      float f = (raw > 0.0f ? 1.0f : 0.0f) * valid[row_i] * valid[b * SQ + j];
      out[(size_t)row_i * SQ + j] = red[v] * f;
    }
  }
}

// ---------------------------------------------------------------------------
extern "C" void kernel_launch(void* const* d_in, const int* in_sizes, int n_in,
                              void* d_out, int out_size, void* d_ws, size_t ws_size,
                              hipStream_t stream) {
  (void)in_sizes; (void)n_in; (void)out_size; (void)ws_size;
  const float* tdm  = (const float*)d_in[0];   // [B,S,S]
  const float* ets  = (const float*)d_in[1];   // [B,S,K]
  const int*   ind  = (const int*)  d_in[2];   // [B,S]
  // d_in[3] = num_types (scalar, unused: K hardcoded)
  const float* w1   = (const float*)d_in[4];   // [H]
  const float* b1   = (const float*)d_in[5];   // [H]
  const float* w2   = (const float*)d_in[6];   // [H, D*D]
  const float* b2   = (const float*)d_in[7];   // [D*D]
  const float* emb1 = (const float*)d_in[8];   // [K+1, D]
  const float* emb2 = (const float*)d_in[9];   // [K+1, D]
  float* out = (float*)d_out;

  // workspace layout
  float*    valid = (float*)d_ws;                      //  1024 f32
  float*    ee1   = valid + BQ * SQ;                   // 16384 f32
  float*    ee2   = ee1 + BQ * SQ * DQ;                // 16384 f32
  _Float16* w2t   = (_Float16*)(ee2 + BQ * SQ * DQ);   // 32768 f16

  prep_kernel<<<(BQ * SQ + 255) / 256, 256, 0, stream>>>(ets, ind, emb1, emb2,
                                                         valid, ee1, ee2);
  w2t_kernel<<<(DQ * DQ * HQ + 255) / 256, 256, 0, stream>>>(w2, w2t);

  const int tiles = BQ * SQ * SQ / 16;                 // 16384
  impact_kernel<<<tiles / 8, 256, 0, stream>>>(tdm, w1, b1, b2,
                                               valid, ee1, ee2, w2t, out);
}
